// MultiHeadAttention_22076131901998
// MI455X (gfx1250) — matrix operand order repaired and checked by reading, not compiled, not run on hardware
//
#include <hip/hip_runtime.h>

// ---------------------------------------------------------------------------
// MHA forward, B=2, S=2048, D=1024, H=16, hd=64 on gfx1250 (MI455X).
// bf16 WMMA, fp32 accum, base-2 flash softmax (scale folded into Q proj),
// K/V staged through double-buffered LDS with global_load_async_to_lds_b128,
// softmax denominator computed on the matrix engine (P x ones).
// ---------------------------------------------------------------------------

typedef __attribute__((ext_vector_type(16))) __bf16 bf16x16;
typedef __attribute__((ext_vector_type(8)))  __bf16 bf16x8;
typedef __attribute__((ext_vector_type(8)))  float  v8f;

#define MODEL_DIM 1024
#define SEQ       2048
#define BATCH     2
#define NHEAD     16
#define HD        64
#define NTOK      (BATCH * SEQ)   // 4096

// 1/sqrt(hd) * log2(e): makes scores base-2 logits directly
#define QSCALE 0.1803368801111204f

__device__ __forceinline__ v8f wmma_bf16(bf16x16 a, bf16x16 b, v8f c) {
  return __builtin_amdgcn_wmma_f32_16x16x32_bf16(
      false, a, false, b, (short)0, c, false, false);
}

__device__ __forceinline__ float hmax16(float v) {
  v = fmaxf(v, __shfl_xor(v, 1));
  v = fmaxf(v, __shfl_xor(v, 2));
  v = fmaxf(v, __shfl_xor(v, 4));
  v = fmaxf(v, __shfl_xor(v, 8));
  return v;
}

// ---------------------------------------------------------------------------
__global__ void __launch_bounds__(256)
cvt_bf16_kernel(const float* __restrict__ in, __bf16* __restrict__ out, int n) {
  int i = (blockIdx.x * 256 + threadIdx.x) * 8;
  if (i >= n) return;
  float4 a = *(const float4*)(in + i);
  float4 b = *(const float4*)(in + i + 4);
  bf16x8 o;
  o[0] = (__bf16)a.x; o[1] = (__bf16)a.y; o[2] = (__bf16)a.z; o[3] = (__bf16)a.w;
  o[4] = (__bf16)b.x; o[5] = (__bf16)b.y; o[6] = (__bf16)b.z; o[7] = (__bf16)b.w;
  *(bf16x8*)(out + i) = o;
}

// ---------------------------------------------------------------------------
// C[M,N] = (A[M,K] * W[N,K]^T + bias) * scale
//   MODE 0: bf16 row-major   MODE 1: bf16 -> V^T [B,H,hd,S]   MODE 2: fp32
// ---------------------------------------------------------------------------
template <int MODE>
__global__ void __launch_bounds__(128)
wmma_gemm_kernel(const __bf16* __restrict__ A, const __bf16* __restrict__ W,
                 const float* __restrict__ bias, void* __restrict__ outp,
                 float scale) {
  const int wave = threadIdx.x >> 5;
  const int lane = threadIdx.x & 31;
  const int L = lane & 15, half = lane >> 4;
  const int m0 = (blockIdx.x * 4 + wave) * 64;
  const int n0 = blockIdx.y * 16;

  const __bf16* aptr = A + (size_t)(m0 + L) * MODEL_DIM + half * 16;
  const __bf16* wptr = W + (size_t)(n0 + L) * MODEL_DIM + half * 16;

  v8f acc[4] = {};

  for (int k0 = 0; k0 < MODEL_DIM; k0 += 32) {
    __builtin_prefetch(wptr + k0 + 128, 0, 1);
    __builtin_prefetch(aptr + k0 + 128, 0, 1);
    bf16x16 bfrag = *(const bf16x16*)(wptr + k0);
    bf16x16 a0 = *(const bf16x16*)(aptr + k0);
    bf16x16 a1 = *(const bf16x16*)(aptr + (size_t)16 * MODEL_DIM + k0);
    bf16x16 a2 = *(const bf16x16*)(aptr + (size_t)32 * MODEL_DIM + k0);
    bf16x16 a3 = *(const bf16x16*)(aptr + (size_t)48 * MODEL_DIM + k0);
    acc[0] = wmma_bf16(a0, bfrag, acc[0]);
    acc[1] = wmma_bf16(a1, bfrag, acc[1]);
    acc[2] = wmma_bf16(a2, bfrag, acc[2]);
    acc[3] = wmma_bf16(a3, bfrag, acc[3]);
  }

  const float bn = bias[n0 + L];

#pragma unroll
  for (int i = 0; i < 4; ++i) {
#pragma unroll
    for (int r = 0; r < 8; ++r) {
      const int row = m0 + i * 16 + r + 8 * half;
      const int col = n0 + L;
      const float v = (acc[i][r] + bn) * scale;
      if (MODE == 0) {
        ((__bf16*)outp)[(size_t)row * MODEL_DIM + col] = (__bf16)v;
      } else if (MODE == 1) {
        const int b = row >> 11, s = row & (SEQ - 1);
        const int h = col >> 6, d = col & (HD - 1);
        ((__bf16*)outp)[((size_t)(b * NHEAD + h) * HD + d) * SEQ + s] = (__bf16)v;
      } else {
        ((float*)outp)[(size_t)row * MODEL_DIM + col] = v;
      }
    }
  }
}

// ---------------------------------------------------------------------------
// Flash attention. Block = 8 waves, all on the same (b,h); wave w owns q-tile
// (base + w). Per 64-key iteration:
//   - 8 waves cooperatively async-copy the next K/V tile (8 KB each) to LDS
//   - 8 score WMMAs from LDS K-tile, base-2 online softmax (max via shuffle
//     butterfly; denominator via P x ones on the matrix engine)
//   - P transposed through wave-private LDS, 8 ctx + 2 lsum WMMAs
// ---------------------------------------------------------------------------
__global__ void __launch_bounds__(256)
flash_attn_kernel(const __bf16* __restrict__ Q, const __bf16* __restrict__ K,
                  const __bf16* __restrict__ Vt, __bf16* __restrict__ Ctx) {
  __shared__ __bf16 kbuf[2][64 * 64];   // [key][d]   8 KB per buffer
  __shared__ __bf16 vbuf[2][64 * 64];   // [d][key]   8 KB per buffer
  __shared__ __bf16 pbuf[8][16 * 64];   // per-wave P transpose scratch

  const int wave = threadIdx.x >> 5;
  const int lane = threadIdx.x & 31;
  const int L = lane & 15, half = lane >> 4;
  const int g = blockIdx.x * 8 + wave;
  const int qt = g & (SEQ / 16 - 1);
  const int h = (g >> 7) & (NHEAD - 1);
  const int b = g >> 11;

  const __bf16* Qbase = Q + (size_t)(b * SEQ + qt * 16) * MODEL_DIM + h * HD;
  const __bf16* Kbase = K + (size_t)(b * SEQ) * MODEL_DIM + h * HD;
  const __bf16* Vbase = Vt + (size_t)((b * NHEAD + h) * HD) * SEQ;
  __bf16* lp = &pbuf[wave][0];

  // cooperative copy roles: each wave moves 8 rows of K and 8 rows of V^T;
  // one async b128 instruction moves 4 rows (8 lanes x 16 B per row).
  const int sub = lane >> 3;
  const int ch = lane & 7;

  auto issue = [&](int j, int bi) {
#pragma unroll
    for (int i = 0; i < 2; ++i) {
      const int row = wave * 8 + i * 4 + sub;  // key index / d index
      unsigned koff = (unsigned)(unsigned long long)(const void*)&kbuf[bi][row * 64 + ch * 8];
      unsigned voff = (unsigned)(unsigned long long)(const void*)&vbuf[bi][row * 64 + ch * 8];
      unsigned long long kg =
          (unsigned long long)(Kbase + (size_t)(j * 64 + row) * MODEL_DIM + ch * 8);
      unsigned long long vg =
          (unsigned long long)(Vbase + (size_t)row * SEQ + j * 64 + ch * 8);
      asm volatile("global_load_async_to_lds_b128 %0, %1, off" ::"v"(koff), "v"(kg)
                   : "memory");
      asm volatile("global_load_async_to_lds_b128 %0, %1, off" ::"v"(voff), "v"(vg)
                   : "memory");
    }
  };

  // Q fragments (16 rows x 64 dims), loaded once (pre-scaled by QSCALE)
  bf16x16 aq0 = *(const bf16x16*)(Qbase + (size_t)L * MODEL_DIM + half * 16);
  bf16x16 aq1 = *(const bf16x16*)(Qbase + (size_t)L * MODEL_DIM + 32 + half * 16);

  // all-ones B fragment: P x ones = row sums, broadcast to all 16 lanes
  bf16x16 ones;
#pragma unroll
  for (int e = 0; e < 16; ++e) ones[e] = (__bf16)1.0f;

  float m[8];
#pragma unroll
  for (int r = 0; r < 8; ++r) m[r] = -3.0e38f;
  v8f ctx[4] = {};
  v8f lsum = {};

  issue(0, 0);

  for (int j = 0; j < SEQ / 64; ++j) {
    const int bi = j & 1;
    if (j + 1 < SEQ / 64) {
      issue(j + 1, bi ^ 1);
      asm volatile("s_wait_asynccnt 4" ::: "memory");  // tile j done (in-order)
    } else {
      asm volatile("s_wait_asynccnt 0" ::: "memory");
    }
    __syncthreads();  // everyone's slice of tile j visible

    const __bf16* kt = &kbuf[bi][0];
    const __bf16* vt = &vbuf[bi][0];

    // ---- scores: 4 key tiles x (2 WMMAs), logits already in log2 domain ----
    v8f s[4];
#pragma unroll
    for (int t = 0; t < 4; ++t) {
      const __bf16* kb = kt + (size_t)(t * 16 + L) * 64 + half * 16;
      bf16x16 b0 = *(const bf16x16*)kb;
      bf16x16 b1 = *(const bf16x16*)(kb + 32);
      v8f acc = {};
      acc = wmma_bf16(aq0, b0, acc);
      acc = wmma_bf16(aq1, b1, acc);
      s[t] = acc;
    }

    // ---- base-2 online softmax; P goes straight to the LDS transpose ----
#pragma unroll
    for (int r = 0; r < 8; ++r) {
      float mx = fmaxf(fmaxf(s[0][r], s[1][r]), fmaxf(s[2][r], s[3][r]));
      mx = hmax16(mx);
      const float mn = fmaxf(m[r], mx);
      const float alpha = exp2f(m[r] - mn);
      m[r] = mn;
      const int row = r + 8 * half;
#pragma unroll
      for (int t = 0; t < 4; ++t)
        lp[row * 64 + t * 16 + L] = (__bf16)exp2f(s[t][r] - mn);
      lsum[r] *= alpha;
      ctx[0][r] *= alpha; ctx[1][r] *= alpha;
      ctx[2][r] *= alpha; ctx[3][r] *= alpha;
    }
    asm volatile("s_wait_dscnt 0" ::: "memory");

    // ---- ctx += P x V, lsum += P x ones : two 32-key P fragments ----
#pragma unroll
    for (int u = 0; u < 2; ++u) {
      bf16x8 plo = *(const bf16x8*)(lp + L * 64 + u * 32 + half * 16);
      bf16x8 phi = *(const bf16x8*)(lp + L * 64 + u * 32 + half * 16 + 8);
      bf16x16 ap;
#pragma unroll
      for (int e = 0; e < 8; ++e) { ap[e] = plo[e]; ap[e + 8] = phi[e]; }
#pragma unroll
      for (int nc = 0; nc < 4; ++nc) {
        const __bf16* vb = vt + (size_t)(nc * 16 + L) * 64 + u * 32 + half * 16;
        bf16x16 bv = *(const bf16x16*)vb;
        ctx[nc] = wmma_bf16(ap, bv, ctx[nc]);
      }
      lsum = wmma_bf16(ap, ones, lsum);  // row sums of P, matrix engine
    }
    __syncthreads();  // all waves done with buffer bi before it is refilled
  }

  // ---- epilogue: normalize, store bf16 ctx [B,S,D] ----
  float inv[8];
#pragma unroll
  for (int r = 0; r < 8; ++r) inv[r] = 1.0f / lsum[r];
  const size_t tok0 = (size_t)(b * SEQ + qt * 16);
#pragma unroll
  for (int nc = 0; nc < 4; ++nc) {
#pragma unroll
    for (int r = 0; r < 8; ++r) {
      const size_t row = tok0 + r + 8 * half;
      const int col = h * HD + nc * 16 + L;
      Ctx[row * MODEL_DIM + col] = (__bf16)(ctx[nc][r] * inv[r]);
    }
  }
}

// ---------------------------------------------------------------------------
extern "C" void kernel_launch(void* const* d_in, const int* in_sizes, int n_in,
                              void* d_out, int out_size, void* d_ws, size_t ws_size,
                              hipStream_t stream) {
  const float* Xq = (const float*)d_in[0];
  const float* Wq = (const float*)d_in[1];
  const float* bq = (const float*)d_in[2];
  const float* Wk = (const float*)d_in[3];
  const float* bk = (const float*)d_in[4];
  const float* Wv = (const float*)d_in[5];
  const float* bv = (const float*)d_in[6];
  const float* Wo = (const float*)d_in[7];
  const float* bo = (const float*)d_in[8];

  char* p = (char*)d_ws;
  __bf16* Xb  = (__bf16*)p; p += (size_t)NTOK * MODEL_DIM * 2;
  __bf16* Wqb = (__bf16*)p; p += (size_t)MODEL_DIM * MODEL_DIM * 2;
  __bf16* Wkb = (__bf16*)p; p += (size_t)MODEL_DIM * MODEL_DIM * 2;
  __bf16* Wvb = (__bf16*)p; p += (size_t)MODEL_DIM * MODEL_DIM * 2;
  __bf16* Wob = (__bf16*)p; p += (size_t)MODEL_DIM * MODEL_DIM * 2;
  __bf16* Qb  = (__bf16*)p; p += (size_t)NTOK * MODEL_DIM * 2;
  __bf16* Kb  = (__bf16*)p; p += (size_t)NTOK * MODEL_DIM * 2;
  __bf16* Vtb = (__bf16*)p; p += (size_t)NTOK * MODEL_DIM * 2;
  __bf16* Cb  = (__bf16*)p; p += (size_t)NTOK * MODEL_DIM * 2;

  const int nX = NTOK * MODEL_DIM;
  const int nW = MODEL_DIM * MODEL_DIM;
  cvt_bf16_kernel<<<nX / 8 / 256, 256, 0, stream>>>(Xq, Xb, nX);
  cvt_bf16_kernel<<<nW / 8 / 256, 256, 0, stream>>>(Wq, Wqb, nW);
  cvt_bf16_kernel<<<nW / 8 / 256, 256, 0, stream>>>(Wk, Wkb, nW);
  cvt_bf16_kernel<<<nW / 8 / 256, 256, 0, stream>>>(Wv, Wvb, nW);
  cvt_bf16_kernel<<<nW / 8 / 256, 256, 0, stream>>>(Wo, Wob, nW);

  dim3 gg(NTOK / 256, MODEL_DIM / 16);
  dim3 gb(128);
  // scale = log2(e)/sqrt(hd) folded into Q so attention works in base-2 logits
  wmma_gemm_kernel<0><<<gg, gb, 0, stream>>>(Xb, Wqb, bq, Qb, QSCALE);
  wmma_gemm_kernel<0><<<gg, gb, 0, stream>>>(Xb, Wkb, bk, Kb, 1.0f);
  wmma_gemm_kernel<1><<<gg, gb, 0, stream>>>(Xb, Wvb, bv, Vtb, 1.0f);

  flash_attn_kernel<<<(BATCH * NHEAD * (SEQ / 16)) / 8, 256, 0, stream>>>(Qb, Kb, Vtb, Cb);

  wmma_gemm_kernel<2><<<gg, gb, 0, stream>>>(Cb, Wob, bo, (void*)d_out, 1.0f);
}
